// LSTMRegressor_69801808495181
// MI455X (gfx1250) — matrix operand order repaired and checked by reading, not compiled, not run on hardware
//
#include <hip/hip_runtime.h>

// ---------------------------------------------------------------------------
// Persistent 2-layer LSTM for MI455X (gfx1250, wave32, WMMA).
//   B=256, T=1024, H0=H1=256, IN=OUT=1, future=0
// 16 persistent workgroups (one per WGP), 256 threads (8 waves) each.
// Each WG owns 16 hidden units (=> 64 gate columns = 4 N-tiles of 16).
// Weights live in LDS as bf16; cell state c lives in VGPRs for all 1024 steps;
// h states ping-pong through bf16 buffers in d_ws; 2 grid barriers / step.
// ---------------------------------------------------------------------------

#define BB 256      // batch
#define TT 1024     // time steps
#define HH 256      // hidden size (both layers)
#define NWG 16      // persistent workgroups
#define HS 16       // hidden units owned per WG
#define NTHREADS 256

typedef __attribute__((ext_vector_type(16))) __bf16        v16bf;
typedef __attribute__((ext_vector_type(8)))  float         v8f;
typedef __attribute__((ext_vector_type(4)))  unsigned int  v4u;

union BfVec {
    v16bf v;
    v4u   u[2];
};

__device__ __forceinline__ float sigmoidf_(float v) {
    return 1.0f / (1.0f + __expf(-v));
}

// Device-wide barrier: monotonically increasing arrival counter in d_ws
// (zeroed by the init kernel each launch, so it is graph-replay safe).
__device__ __forceinline__ void gridbar(unsigned* bar, unsigned target) {
    __syncthreads();
    if (threadIdx.x == 0) {
        __threadfence();
        atomicAdd(bar, 1u);
        while (__hip_atomic_load(bar, __ATOMIC_ACQUIRE, __HIP_MEMORY_SCOPE_AGENT) < target) {
            __builtin_amdgcn_s_sleep(1);
        }
    }
    __syncthreads();
}

// One K=256 GEMM pass: acc[2][4] += A(h rows) x B(weight slice in LDS).
// A: 16x32 bf16 tiles loaded from row-major h buffer (2x b128 per lane).
// B: 32x16 bf16 tiles; W row-major == B column-major, contiguous per lane.
__device__ __forceinline__ void mm_pass(const __bf16* __restrict__ hbuf,
                                        const __bf16* __restrict__ Ws,
                                        v8f (&acc)[2][4],
                                        int wv, int nloc, int hl)
{
    const int row0 = (2 * wv) * 16 + nloc;   // A-tile row for this lane
    const int row1 = row0 + 16;
#pragma unroll
    for (int kk = 0; kk < 8; ++kk) {
        const int ka = kk * 32 + hl * 8;     // A per-lane K base (ISA layout)
        BfVec a0, a1;
        a0.u[0] = *(const v4u*)(hbuf + row0 * HH + ka);
        a0.u[1] = *(const v4u*)(hbuf + row0 * HH + ka + 16);
        a1.u[0] = *(const v4u*)(hbuf + row1 * HH + ka);
        a1.u[1] = *(const v4u*)(hbuf + row1 * HH + ka + 16);
        const int kb = kk * 32 + hl * 8;     // B per-lane K base (ISA layout)
#pragma unroll
        for (int g = 0; g < 4; ++g) {
            BfVec bb;
            const __bf16* bp = Ws + (g * 16 + nloc) * HH + kb;
            bb.u[0] = *(const v4u*)(bp);
            bb.u[1] = *(const v4u*)(bp + 16);
            acc[0][g] = __builtin_amdgcn_wmma_f32_16x16x32_bf16(
                false, a0.v, false, bb.v, (short)0, acc[0][g], false, false);
            acc[1][g] = __builtin_amdgcn_wmma_f32_16x16x32_bf16(
                false, a1.v, false, bb.v, (short)0, acc[1][g], false, false);
        }
    }
}

// Per-launch init (stream-ordered before the persistent kernel):
// zero barrier counter, zero h ping-pong buffer 0, seed out with b_lin.
__global__ void lstm_init(unsigned* bar, __bf16* h0g, __bf16* h1g,
                          float* out, const float* blin)
{
    const int i = blockIdx.x * blockDim.x + threadIdx.x;
    const int stride = gridDim.x * blockDim.x;
    if (i == 0) *bar = 0u;
    const float bl = blin[0];
    for (int j = i; j < BB * HH; j += stride) {
        h0g[j] = (__bf16)0.0f;
        h1g[j] = (__bf16)0.0f;
    }
    for (int j = i; j < BB * TT; j += stride) out[j] = bl;
}

__global__ void __launch_bounds__(NTHREADS, 1)
lstm_persistent(const float* __restrict__ x,
                const float* __restrict__ Wih0, const float* __restrict__ Whh0,
                const float* __restrict__ bih0, const float* __restrict__ bhh0,
                const float* __restrict__ Wih1, const float* __restrict__ Whh1,
                const float* __restrict__ bih1, const float* __restrict__ bhh1,
                const float* __restrict__ Wlin,
                float* __restrict__ out,
                unsigned* __restrict__ bar,
                __bf16* __restrict__ h0g,     // [2][BB][HH] ping-pong
                __bf16* __restrict__ h1g)     // [2][BB][HH] ping-pong
{
    extern __shared__ char smem_raw[];
    __bf16* Whh0_s = (__bf16*)smem_raw;            // [64][256] bf16 = 32KB
    __bf16* Wih1_s = Whh0_s + 64 * HH;             // 32KB
    __bf16* Whh1_s = Wih1_s + 64 * HH;             // 32KB
    float*  xs     = (float*)(Whh1_s + 64 * HH);   // [256] x_t column

    const int tid    = threadIdx.x;
    const int wg     = blockIdx.x;      // 0..15
    const int lane   = tid & 31;
    const int wv     = tid >> 5;        // wave 0..7
    const int nloc   = lane & 15;       // local gate column / B-tile column
    const int hl     = lane >> 4;       // half-wave select
    const int wgbase = wg * HS;         // first hidden unit owned by this WG

    // ---- stage this WG's weight slices into LDS as bf16 (once) ----
    for (int idx = tid; idx < 64 * HH; idx += NTHREADS) {
        const int r = idx >> 8;         // 0..63: local gate row
        const int k = idx & (HH - 1);
        const int g = r >> 4;           // gate 0..3 (i,f,g,o)
        const int j = r & 15;           // local hidden unit
        const int grow = g * HH + wgbase + j;   // row in [4H][H] weight
        Whh0_s[idx] = (__bf16)Whh0[grow * HH + k];
        Wih1_s[idx] = (__bf16)Wih1[grow * HH + k];
        Whh1_s[idx] = (__bf16)Whh1[grow * HH + k];
    }

    // per-lane constants: IN==1 so x-contribution is x[b]*Wih0[n]; fused biases
    float wih0v[4], bias0v[4], bias1v[4];
#pragma unroll
    for (int g = 0; g < 4; ++g) {
        const int n = g * HH + wgbase + nloc;
        wih0v[g]  = Wih0[n];
        bias0v[g] = bih0[n] + bhh0[n];
        bias1v[g] = bih1[n] + bhh1[n];
    }
    const float wlinv = Wlin[wgbase + nloc];   // OUT==1

    // cell states live in VGPRs for the whole sequence (2 tiles x 8 rows)
    float c0r[16], c1r[16];
#pragma unroll
    for (int i = 0; i < 16; ++i) { c0r[i] = 0.0f; c1r[i] = 0.0f; }

    __syncthreads();

    unsigned bc = 0;

    for (int t = 0; t < TT; ++t) {
        const int rd = t & 1;
        const int wr = rd ^ 1;
        const __bf16* h0rd = h0g + rd * BB * HH;
        __bf16*       h0wr = h0g + wr * BB * HH;
        const __bf16* h1rd = h1g + rd * BB * HH;
        __bf16*       h1wr = h1g + wr * BB * HH;

        // stage x[:, t] into LDS (thread == batch row)
        xs[tid] = x[tid * TT + t];
        __syncthreads();

        // ================= layer 0: gates = h0 @ Whh0^T (+x, +bias) ========
        v8f acc[2][4];
#pragma unroll
        for (int m = 0; m < 2; ++m)
#pragma unroll
            for (int g = 0; g < 4; ++g)
#pragma unroll
                for (int i = 0; i < 8; ++i) acc[m][g][i] = 0.0f;

        mm_pass(h0rd, Whh0_s, acc, wv, nloc, hl);

#pragma unroll
        for (int mt = 0; mt < 2; ++mt) {
            const int bbase = (2 * wv + mt) * 16 + 8 * hl;
#pragma unroll
            for (int r = 0; r < 8; ++r) {
                const int b = bbase + r;
                const float xv = xs[b];
                const float gi = acc[mt][0][r] + xv * wih0v[0] + bias0v[0];
                const float gf = acc[mt][1][r] + xv * wih0v[1] + bias0v[1];
                const float gc = acc[mt][2][r] + xv * wih0v[2] + bias0v[2];
                const float go = acc[mt][3][r] + xv * wih0v[3] + bias0v[3];
                const float c  = sigmoidf_(gf) * c0r[mt * 8 + r]
                               + sigmoidf_(gi) * tanhf(gc);
                const float h  = sigmoidf_(go) * tanhf(c);
                c0r[mt * 8 + r] = c;
                h0wr[b * HH + wgbase + nloc] = (__bf16)h;
            }
        }

        gridbar(bar, (++bc) * NWG);   // all of h0(t) visible

        // ===== layer 1: gates = h0_new @ Wih1^T + h1 @ Whh1^T (+bias) ======
#pragma unroll
        for (int m = 0; m < 2; ++m)
#pragma unroll
            for (int g = 0; g < 4; ++g)
#pragma unroll
                for (int i = 0; i < 8; ++i) acc[m][g][i] = 0.0f;

        mm_pass(h0wr, Wih1_s, acc, wv, nloc, hl);
        mm_pass(h1rd, Whh1_s, acc, wv, nloc, hl);

#pragma unroll
        for (int mt = 0; mt < 2; ++mt) {
            const int bbase = (2 * wv + mt) * 16 + 8 * hl;
#pragma unroll
            for (int r = 0; r < 8; ++r) {
                const int b = bbase + r;
                const float gi = acc[mt][0][r] + bias1v[0];
                const float gf = acc[mt][1][r] + bias1v[1];
                const float gc = acc[mt][2][r] + bias1v[2];
                const float go = acc[mt][3][r] + bias1v[3];
                const float c  = sigmoidf_(gf) * c1r[mt * 8 + r]
                               + sigmoidf_(gi) * tanhf(gc);
                const float h  = sigmoidf_(go) * tanhf(c);
                c1r[mt * 8 + r] = c;
                h1wr[b * HH + wgbase + nloc] = (__bf16)h;

                // out[b,t] += sum over this WG's 16 hidden units of h*Wlin
                float v = h * wlinv;
                v += __shfl_xor(v, 1, 32);
                v += __shfl_xor(v, 2, 32);
                v += __shfl_xor(v, 4, 32);
                v += __shfl_xor(v, 8, 32);
                if (nloc == 0) atomicAdd(&out[b * TT + t], v);
            }
        }

        gridbar(bar, (++bc) * NWG);   // all of h1(t) visible for t+1
    }
}

extern "C" void kernel_launch(void* const* d_in, const int* in_sizes, int n_in,
                              void* d_out, int out_size, void* d_ws, size_t ws_size,
                              hipStream_t stream) {
    (void)in_sizes; (void)n_in; (void)out_size; (void)ws_size;
    const float* x    = (const float*)d_in[0];
    const float* Wih0 = (const float*)d_in[1];
    const float* Whh0 = (const float*)d_in[2];
    const float* bih0 = (const float*)d_in[3];
    const float* bhh0 = (const float*)d_in[4];
    const float* Wih1 = (const float*)d_in[5];
    const float* Whh1 = (const float*)d_in[6];
    const float* bih1 = (const float*)d_in[7];
    const float* bhh1 = (const float*)d_in[8];
    const float* Wlin = (const float*)d_in[9];
    const float* blin = (const float*)d_in[10];
    // d_in[11] = future (==0 in this benchmark)

    float* out = (float*)d_out;

    // workspace layout: [barrier counter | pad to 256B | h0 ping-pong | h1 ping-pong]
    char* ws = (char*)d_ws;
    unsigned* bar = (unsigned*)ws;
    __bf16* h0g = (__bf16*)(ws + 256);
    __bf16* h1g = h0g + 2 * BB * HH;

    lstm_init<<<64, 256, 0, stream>>>(bar, h0g, h1g, out, blin);

    const size_t smem = (size_t)(3 * 64 * HH) * sizeof(__bf16)   // 96KB weights
                      + (size_t)256 * sizeof(float);             // x_t column
    lstm_persistent<<<NWG, NTHREADS, smem, stream>>>(
        x, Wih0, Whh0, bih0, bhh0, Wih1, Whh1, bih1, bhh1, Wlin,
        out, bar, h0g, h1g);
}
